// CustomFourierLayer_45243185496812
// MI455X (gfx1250) — matrix operand order repaired and checked by reading, not compiled, not run on hardware
//
#include <hip/hip_runtime.h>

// Problem dims (fixed by the reference)
#define BROWS 4096            // batch B
#define IDIM  1024            // input dim I
#define ODIM  1024            // output dim O
#define FPER  33              // 2*16+1 features per input
#define KTOT  (IDIM * FPER)   // 33792 = flattened K
#define TM    128             // M tile per workgroup
#define TN    128             // N tile per workgroup
#define KSTEP 32              // WMMA K step (f16 16x16x32)
#define NKSTEPS (KTOT / KSTEP) // 1056 = 32 iblocks * 33 r-steps
#define THREADS 256           // 8 wave32
#define WS_ELEMS ((size_t)ODIM * (size_t)KTOT)   // fp16 coef copy element count

typedef __attribute__((ext_vector_type(16))) _Float16 v16h;
typedef __attribute__((ext_vector_type(8)))  float    v8f;
typedef __attribute__((ext_vector_type(4)))  unsigned u32x4;
typedef __attribute__((ext_vector_type(8)))  unsigned u32x8;

__device__ __forceinline__ unsigned short f2h(float f) {
    _Float16 h = (_Float16)f;
    return __builtin_bit_cast(unsigned short, h);
}

// Generic pointers to LDS carry the workgroup-relative LDS offset in the low
// 32 bits (the flat shared aperture base sits in the high half).
__device__ __forceinline__ unsigned lds_off(const void* p) {
    return (unsigned)(size_t)p;
}

// Async DMA: 16 bytes per lane, global -> LDS, tracked by ASYNCcnt.
__device__ __forceinline__ void async_load_b128(unsigned ldsdst, const void* src) {
    asm volatile("global_load_async_to_lds_b128 %0, %1, off"
                 :: "v"(ldsdst), "v"(src)
                 : "memory");
}

__device__ __forceinline__ void wait_asynccnt0() {
#if __has_builtin(__builtin_amdgcn_s_wait_asynccnt)
    __builtin_amdgcn_s_wait_asynccnt(0);
#else
    asm volatile("s_wait_asynccnt 0x0" ::: "memory");
#endif
}

__device__ __forceinline__ void wait_tensorcnt0() {
#if __has_builtin(__builtin_amdgcn_s_wait_tensorcnt)
    __builtin_amdgcn_s_wait_tensorcnt(0);
#else
    asm volatile("s_wait_tensorcnt 0x0" ::: "memory");
#endif
}

// TDM: one instruction DMAs a 2D tile (tile_w f16 x tile_h rows) from global
// into LDS. D# per CDNA5 ISA ch.8: group0 = {count, lds_addr, global_addr,
// type=2}; group1 = {mask=0, data_size, dims, tile dims, dim0 stride}.
__device__ __forceinline__ void tdm_load_2d_f16(unsigned lds_byte,
                                                const void* gaddr,
                                                unsigned tile_w, unsigned tile_h,
                                                unsigned stride_elems) {
    const unsigned long long ga = (unsigned long long)gaddr;
    u32x4 g0;
    g0[0] = 1u;                                           // count=1 (valid), user
    g0[1] = lds_byte;                                     // lds_addr [63:32]
    g0[2] = (unsigned)ga;                                 // global_addr [95:64]
    g0[3] = (unsigned)((ga >> 32) & 0x01FFFFFFu) | (2u << 30); // [120:96] | type=2
    u32x8 g1;
    g1[0] = (1u << 16);                                   // data_size=1 (2 bytes)
    g1[1] = (33792u & 0xFFFFu) << 16;                     // tensor_dim0 lo16 @63:48
    g1[2] = (33792u >> 16) | ((1024u & 0xFFFFu) << 16);   // dim0 hi @79:64, dim1 lo @95:80
    g1[3] = (1024u >> 16) | (tile_w << 16);               // dim1 hi @111:96, tile_dim0 @127:112
    g1[4] = tile_h & 0xFFFFu;                             // tile_dim1 @143:128, tile_dim2=0
    g1[5] = stride_elems;                                 // tensor_dim0_stride lo32 @191:160
    g1[6] = 0u;                                           // stride hi16, dim1_stride lo16
    g1[7] = 0u;
    asm volatile("tensor_load_to_lds %0, %1" :: "s"(g0), "s"(g1) : "memory");
}

union Frag16 {
    v16h v;
    struct { uint4 lo, hi; } q;
};

// ---------------------------------------------------------------------------
// Kernel 1: coef f32 -> f16 into workspace, PERMUTED K order:
//   kk' = iblock*1056 + r*32 + ii, i = iblock*32+ii, feature order interleaved
//   [sin x, cos x, ..., sin 16x, cos 16x, 1]. Makes A-side gen a recurrence.
// One-shot ~138MB read + 69MB write (~9us); fp16 coef (69MB) then L2-resident.
// ---------------------------------------------------------------------------
__global__ __launch_bounds__(256)
void coef_permute_f16(const float* __restrict__ coef, unsigned short* __restrict__ ws) {
    const int o = blockIdx.x;                        // output row 0..1023
    const float* src = coef + (size_t)o * KTOT;
    unsigned short* dst = ws + (size_t)o * KTOT;
    for (int ib = 0; ib < 32; ++ib) {
        for (int e = threadIdx.x; e < 1056; e += 256) {
            const int r  = e >> 5;                   // 0..32
            const int ii = e & 31;
            const int i  = ib * 32 + ii;
            const int ko = (r == 32) ? 32 : ((r & 1) ? 16 + (r >> 1) : (r >> 1));
            dst[ib * 1056 + e] = f2h(src[i * 33 + ko]);
        }
    }
}

// ---------------------------------------------------------------------------
// Kernel 2: fused feature-gen + GEMM.
//   out[4096,1024] = V[4096,33792] * W[1024,33792]^T  (shared K permutation)
// MODE: 0 = no-ws fallback (f32 coef, direct sin/cos), 1 = per-lane async DMA,
//       2 = TDM tensor_load_to_lds (one 8KB tile DMA per K-step).
// ---------------------------------------------------------------------------
template <int MODE>
__global__ __launch_bounds__(THREADS)
void fourier_wmma_f16(const float* __restrict__ x,
                      const float* __restrict__ coef,
                      const unsigned short* __restrict__ wcoef,
                      float* __restrict__ out)
{
    __shared__ unsigned short At[2][TM * KSTEP];   // A[row][kk], 8KB each
    __shared__ unsigned short Bt[2][TN * KSTEP];   // B[ocol][kk], 8KB each

    const int mbase = blockIdx.x * TM;
    const int nbase = blockIdx.y * TN;

    const int tid  = threadIdx.x;
    const int wave = tid >> 5;
    const int lane = tid & 31;
    const int half = lane >> 4;
    const int l16  = lane & 15;

    // Staging: thread owns row rs, 16-element (32B) K segment seg
    const int rs  = tid >> 1;            // 0..127
    const int seg = (tid & 1) * 16;      // 0 or 16 (f16 elements)
    const float*          xrow  = x     + (size_t)(mbase + rs) * IDIM;
    const float*          crow  = coef  + (size_t)(nbase + rs) * KTOT;
    const unsigned short* wrow  = wcoef + (size_t)(nbase + rs) * KTOT;
    const unsigned short* wtile = wcoef + (size_t)nbase * KTOT;   // TDM tile base

    v8f acc[8] = {};

    // Recurrence state: sin/cos of k*x and of x for this thread's 16 fixed
    // inputs within the current iblock (fully unrolled -> registers).
    float ss[16], cc[16], s1v[16], c1v[16];

    auto stage = [&](int buf, int stepIdx) {
        const int k0 = stepIdx * KSTEP;
        // ---- B tile: 128x32 f16
        if constexpr (MODE == 2) {
            if (tid < 32)  // one wave issues the tile DMA; TDM ignores EXEC
                tdm_load_2d_f16(lds_off(&Bt[buf][0]), wtile + k0,
                                KSTEP, TN, KTOT);
        } else if constexpr (MODE == 1) {
            const unsigned short* src = wrow + k0 + seg;
            unsigned d = lds_off(&Bt[buf][rs * KSTEP + seg]);
            async_load_b128(d,      src);
            async_load_b128(d + 16, src + 8);
        } else {
            const float4* src = (const float4*)(crow + k0 + seg); // 64B aligned
            float4 f4[4];
            #pragma unroll
            for (int q = 0; q < 4; ++q) f4[q] = src[q];
            const float* fs = (const float*)f4;
            unsigned short tmp[16];
            #pragma unroll
            for (int c = 0; c < 16; ++c) tmp[c] = f2h(fs[c]);
            uint4* d = (uint4*)&Bt[buf][rs * KSTEP + seg];
            d[0] = ((const uint4*)tmp)[0];
            d[1] = ((const uint4*)tmp)[1];
        }
        // ---- A tile: Fourier features -> f16
        unsigned short tmp[16];
        if constexpr (MODE >= 1) {
            // step = (iblock, r); thread's 16 inputs constant for 33 steps.
            const int r = stepIdx % 33;              // uniform -> scalar branch
            if (r == 0) {
                const float* xp = xrow + (stepIdx / 33) * 32 + seg;
                #pragma unroll
                for (int c = 0; c < 16; ++c) {
                    float s, co;
                    __sincosf(xp[c], &s, &co);
                    s1v[c] = s; c1v[c] = co;
                    ss[c] = s; cc[c] = co;
                    tmp[c] = f2h(s);                 // sin(1x)
                }
            } else if (r == 32) {
                #pragma unroll
                for (int c = 0; c < 16; ++c) tmp[c] = f2h(1.0f);
            } else if (r & 1) {
                #pragma unroll
                for (int c = 0; c < 16; ++c) {
                    tmp[c] = f2h(cc[c]);             // cos(kx)
                    const float ns = ss[c] * c1v[c] + cc[c] * s1v[c];
                    const float nc = cc[c] * c1v[c] - ss[c] * s1v[c];
                    ss[c] = ns; cc[c] = nc;          // k -> k+1
                }
            } else {
                #pragma unroll
                for (int c = 0; c < 16; ++c) tmp[c] = f2h(ss[c]);  // sin(kx)
            }
        } else {
            // Fallback: coef native order, direct per-element sin/cos
            #pragma unroll
            for (int c = 0; c < 16; ++c) {
                unsigned kk = (unsigned)(k0 + seg + c);
                unsigned i  = kk / 33u;
                unsigned r  = kk - i * 33u;
                float xv = xrow[i];
                float f;
                if (r < 16u)      f = __sinf(xv * (float)(r + 1u));
                else if (r < 32u) f = __cosf(xv * (float)(r - 15u));
                else              f = 1.0f;
                tmp[c] = f2h(f);
            }
        }
        uint4* d = (uint4*)&At[buf][rs * KSTEP + seg];
        d[0] = ((const uint4*)tmp)[0];
        d[1] = ((const uint4*)tmp)[1];
    };

    auto wait_dma = [&]() {
        if constexpr (MODE == 2) wait_tensorcnt0();
        else if constexpr (MODE == 1) wait_asynccnt0();
    };

    stage(0, 0);
    wait_dma();
    __syncthreads();

    const int arow = wave * 16 + l16;  // A-fragment row for this lane

    for (int step = 0; step < NKSTEPS; ++step) {
        const int cur = step & 1;
        if (step + 1 < NKSTEPS) {
            if constexpr (MODE == 0) {
                if (step + 2 < NKSTEPS)
                    __builtin_prefetch(crow + (step + 2) * KSTEP + seg, 0, 1);
            }
            stage(cur ^ 1, step + 1);
        }

        // A fragment: 16 contiguous K at half*16 (this kernel's K order, same as B)
        Frag16 a;
        a.q.lo = *(const uint4*)&At[cur][arow * KSTEP + half * 16];
        a.q.hi = *(const uint4*)&At[cur][arow * KSTEP + half * 16 + 8];

        #pragma unroll
        for (int t = 0; t < 8; ++t) {
            // B fragment: lane = col, 16 contiguous K at half*16 (same K order as A)
            Frag16 b;
            const int bcol = t * 16 + l16;
            b.q.lo = *(const uint4*)&Bt[cur][bcol * KSTEP + half * 16];
            b.q.hi = *(const uint4*)&Bt[cur][bcol * KSTEP + half * 16 + 8];
            acc[t] = __builtin_amdgcn_wmma_f32_16x16x32_f16(
                false, a.v, false, b.v, (short)0, acc[t], false, false);
        }

        wait_dma();
        __syncthreads();
    }

    // Epilogue: C/D layout -> row = r + half*8, col = l16 (f32 output)
    const int orow0 = mbase + wave * 16 + half * 8;
    #pragma unroll
    for (int t = 0; t < 8; ++t) {
        const int ocol = nbase + t * 16 + l16;
        #pragma unroll
        for (int r = 0; r < 8; ++r) {
            out[(size_t)(orow0 + r) * ODIM + ocol] = acc[t][r];
        }
    }
}

// Keep the per-lane async variant compiled into the binary as well.
template __global__ void fourier_wmma_f16<1>(const float* __restrict__,
                                             const float* __restrict__,
                                             const unsigned short* __restrict__,
                                             float* __restrict__);

extern "C" void kernel_launch(void* const* d_in, const int* in_sizes, int n_in,
                              void* d_out, int out_size, void* d_ws, size_t ws_size,
                              hipStream_t stream) {
    const float* x    = (const float*)d_in[0];   // [4096, 1024] f32
    const float* coef = (const float*)d_in[1];   // [1024, 1024, 33] f32
    float* out = (float*)d_out;                  // [4096, 1024] f32
    (void)in_sizes; (void)n_in; (void)out_size;

    dim3 grid(BROWS / TM, ODIM / TN);            // 32 x 8 = 256 workgroups

    if (ws_size >= WS_ELEMS * sizeof(unsigned short)) {
        unsigned short* w = (unsigned short*)d_ws;
        coef_permute_f16<<<ODIM, 256, 0, stream>>>(coef, w);
        fourier_wmma_f16<2><<<grid, THREADS, 0, stream>>>(x, coef, w, out);
    } else {
        fourier_wmma_f16<0><<<grid, THREADS, 0, stream>>>(x, coef, nullptr, out);
    }
}
